// LSTM_CRF_13752485282004
// MI455X (gfx1250) — matrix operand order repaired and checked by reading, not compiled, not run on hardware
//
#include <hip/hip_runtime.h>
#include <hip/hip_bf16.h>
#include <math.h>

// Problem dims
#define B_   32
#define T_   512
#define E_   768
#define H_   256
#define K_   5
#define G4H  1024   // 4*H
#define BT   (B_*T_)
#define NT_  64     // N tiles across 4H (1024/16)

typedef __attribute__((ext_vector_type(16))) __bf16          v16bf;
typedef __attribute__((ext_vector_type(16))) unsigned short  v16u;
typedef __attribute__((ext_vector_type(8)))  float           v8f;
typedef __attribute__((ext_vector_type(4)))  unsigned int    v4u;
typedef __attribute__((ext_vector_type(8)))  unsigned int    v8u;

// ---- helpers ---------------------------------------------------------------

__device__ __forceinline__ unsigned short f2bf(float x) {
  unsigned int u = __builtin_bit_cast(unsigned int, x);
  unsigned int r = (u + 0x7FFFu + ((u >> 16) & 1u)) >> 16;  // RNE
  return (unsigned short)r;
}

__device__ __forceinline__ float sigm(float x) { return 1.0f / (1.0f + expf(-x)); }

// A fragment: 16x32 bf16 (M rows x K cols), CDNA5 layout:
// lane: half=l>>4, m=l&15; element j: k = 16*(j>=8) + 8*half + (j&7)
// -> j=0..7 and j=8..15 are each contiguous 16B runs => 2x b128 loads.
__device__ __forceinline__ v16bf a_frag(const unsigned short* base, int ld) {
  int lane = threadIdx.x & 31;
  int half = lane >> 4, m = lane & 15;
  v16u u;
#pragma unroll
  for (int j = 0; j < 16; ++j) {
    int k = ((j >> 3) << 4) + (half << 3) + (j & 7);
    u[j] = base[m * ld + k];
  }
  return __builtin_bit_cast(v16bf, u);
}

// B fragment from pre-swizzled weights: tile stored as 32 lanes x 16 bf16
// contiguous (1KB per 32x16 tile) -> one 32B contiguous load per lane.
__device__ __forceinline__ v16bf b_frag_swz(const unsigned short* tile_base) {
  int lane = threadIdx.x & 31;
  const v16u* p = (const v16u*)tile_base + lane;
  return __builtin_bit_cast(v16bf, *p);
}

// TDM: 2-D tile load Global -> LDS (CDNA5 ISA).
// Tile: 16 rows x 64 bytes (8-byte units), row stride E_*2 bytes.
// Issued from one wave; tracked with TENSORcnt.
__device__ __forceinline__ void tdm_load_tile16x64(unsigned lds_byte,
                                                   unsigned long long gaddr) {
  v4u d0;
  d0[0] = 1u;                                   // count=1 valid, user D#
  d0[1] = lds_byte;                             // lds_addr (bytes)
  d0[2] = (unsigned)(gaddr & 0xFFFFFFFFu);      // global_addr[31:0]
  d0[3] = ((unsigned)(gaddr >> 32) & 0x01FFFFFFu) | 0x80000000u;  // [56:32], type=2
  v8u d1;
  d1[0] = 3u << 16;                             // wg_mask=0, data_size=3 (8B)
  d1[1] = 8u << 16;                             // tensor_dim0 = 8 units (64B)
  d1[2] = 16u << 16;                            // tensor_dim1 = 16 rows
  d1[3] = 8u << 16;                             // tile_dim0 = 8 units
  d1[4] = 16u;                                  // tile_dim1 = 16, tile_dim2 = 0
  d1[5] = (unsigned)(E_ * 2 / 8);               // tensor_dim0_stride = 192 units
  d1[6] = 0u;
  d1[7] = 0u;
  asm volatile("tensor_load_to_lds %0, %1" :: "s"(d0), "s"(d1) : "memory");
}

// ---- K0: pack weights into WMMA-fragment order (bf16) ----------------------
// swz[((kt*NT + nt)*32 + lane)*16 + j] = W[g][e],
//   g = nt*16 + (lane&15),  e = kt*32 + 16*(j>>3) + 8*(lane>>4) + (j&7)

__global__ void __launch_bounds__(256) pack_weights_kernel(
    const float* __restrict__ wihf, const float* __restrict__ wihb,
    const float* __restrict__ whhf, const float* __restrict__ whhb,
    unsigned short* __restrict__ wihf_s, unsigned short* __restrict__ wihb_s,
    unsigned short* __restrict__ whhf_s, unsigned short* __restrict__ whhb_s) {
  int o = blockIdx.x * blockDim.x + threadIdx.x;   // over 4H*E = 786432
  if (o >= G4H * E_) return;
  int j  = o & 15;
  int l  = (o >> 4) & 31;
  int nt = (o >> 9) & 63;
  int kt = o >> 15;
  int g = nt * 16 + (l & 15);
  int e = kt * 32 + ((j >> 3) << 4) + ((l >> 4) << 3) + (j & 7);
  wihf_s[o] = f2bf(wihf[g * E_ + e]);
  wihb_s[o] = f2bf(wihb[g * E_ + e]);
  if (o < G4H * H_) {             // kt < 8  =>  e < 256
    whhf_s[o] = f2bf(whhf[g * H_ + e]);
    whhb_s[o] = f2bf(whhb[g * H_ + e]);
  }
}

// ---- K1: embeddings fp32 -> bf16 (forward + length-reversed) ---------------

__global__ void __launch_bounds__(256) pack_embs_kernel(
    const float* __restrict__ we, const int* __restrict__ lengths,
    unsigned short* __restrict__ ef, unsigned short* __restrict__ er) {
  size_t idx = (size_t)blockIdx.x * blockDim.x + threadIdx.x;
  if (idx >= (size_t)BT * E_) return;
  int e  = (int)(idx % E_);
  int bt = (int)(idx / E_);
  int t  = bt % T_;
  int b  = bt / T_;
  ef[idx] = f2bf(we[idx]);
  int len = lengths[b];
  int ts = (t < len) ? (len - 1 - t) : t;
  er[idx] = f2bf(we[((size_t)b * T_ + ts) * E_ + e]);
}

// ---- K2: xg = embs @ W_ih^T + b  (bf16 WMMA, f32 accum) --------------------
// grid (BT/16, 2, 2dirs), block 256 (8 waves).
// Block covers M=16 x N=512; each wave owns 4 consecutive N-tiles.
// A tile (16x32 bf16 = 1KB) staged to LDS by the Tensor Data Mover,
// double-buffered: tile kt+1 DMAs while tile kt feeds the WMMAs.

__global__ void __launch_bounds__(256) xg_gemm_kernel(
    const unsigned short* __restrict__ embs_f, const unsigned short* __restrict__ embs_r,
    const unsigned short* __restrict__ wih_f_s, const unsigned short* __restrict__ wih_b_s,
    const float* __restrict__ bias_f, const float* __restrict__ bias_b,
    float* __restrict__ xg_f, float* __restrict__ xg_b) {
  const int dir = blockIdx.z;
  const unsigned short* embs = dir ? embs_r : embs_f;
  const unsigned short* wsz  = dir ? wih_b_s : wih_f_s;
  const float*          bias = dir ? bias_b  : bias_f;
  float*                xg   = dir ? xg_b    : xg_f;

  const int m0   = blockIdx.x * 16;
  const int wave = threadIdx.x >> 5;
  const int lane = threadIdx.x & 31;
  const int nt0  = blockIdx.y * 32 + wave * 4;   // first of 4 N-tiles

  __shared__ __align__(16) unsigned short As[2][16 * 32];
  const unsigned lds0 = (unsigned)(size_t)&As[0][0];
  const unsigned lds1 = (unsigned)(size_t)&As[1][0];
  const unsigned long long gbase =
      (unsigned long long)(size_t)(embs + (size_t)m0 * E_);

  v8f acc[4];
#pragma unroll
  for (int q = 0; q < 4; ++q) acc[q] = (v8f){0.f,0.f,0.f,0.f,0.f,0.f,0.f,0.f};

  if (threadIdx.x < 32) tdm_load_tile16x64(lds0, gbase);   // prologue: tile 0

  for (int kt = 0; kt < E_ / 32; ++kt) {
    __syncthreads();                 // compute(kt-1) done; safe to refill
    if (threadIdx.x < 32) {
      if (kt + 1 < E_ / 32) {
        tdm_load_tile16x64(((kt + 1) & 1) ? lds1 : lds0,
                           gbase + (unsigned long long)(kt + 1) * 64ull);
        asm volatile("s_wait_tensorcnt 0x1" ::: "memory");  // tile kt landed
      } else {
        asm volatile("s_wait_tensorcnt 0x0" ::: "memory");
      }
    }
    if (kt + 1 < E_ / 32)            // pull next B chunk toward L2/L0
      __builtin_prefetch(wsz + ((size_t)(kt + 1) * NT_ + nt0) * 512, 0, 1);
    __syncthreads();                 // tile kt visible to all waves

    v16bf a = a_frag(As[kt & 1], 32);
    const unsigned short* wb = wsz + ((size_t)kt * NT_ + nt0) * 512;
#pragma unroll
    for (int q = 0; q < 4; ++q) {
      v16bf b = b_frag_swz(wb + (size_t)q * 512);
      acc[q] = __builtin_amdgcn_wmma_f32_16x16x32_bf16(false, a, false, b,
                                                       (short)0, acc[q], false, false);
    }
  }

  const int half = lane >> 4, n = lane & 15;
#pragma unroll
  for (int q = 0; q < 4; ++q) {
#pragma unroll
    for (int r = 0; r < 8; ++r) {
      int row = m0 + r + 8 * half;
      int col = (nt0 + q) * 16 + n;
      xg[(size_t)row * G4H + col] = acc[q][r] + bias[col];
    }
  }
}

// ---- K3: recurrent LSTM, one 32-wave workgroup per direction ---------------
// h (bf16) and c (f32) live in static LDS; xg[t] slabs (32x1024 f32 = 128KB)
// are double-buffered in dynamic LDS via global_load_async_to_lds_b128 so the
// next step's gate inputs stream in behind the current step's WMMAs.
// Total LDS: 2*128KB + 48KB = 304KB (<= 320KB WGP budget, 1 WG per WGP).

__global__ void __launch_bounds__(1024) lstm_kernel(
    const float* __restrict__ xg_f, const float* __restrict__ xg_b,
    const unsigned short* __restrict__ whh_f_s, const unsigned short* __restrict__ whh_b_s,
    const int* __restrict__ lengths,
    float* __restrict__ hf, float* __restrict__ hb) {
  const int dir = blockIdx.x;
  const float*          xg  = dir ? xg_b : xg_f;
  const unsigned short* wsz = dir ? whh_b_s : whh_f_s;
  float*              hout  = dir ? hb : hf;

  extern __shared__ __align__(16) char dynls[];
  float* xg_s = (float*)dynls;                 // 2 x 32*1024 floats
  __shared__ unsigned short h_s[32 * 256];
  __shared__ float          c_s[32 * 256];

  const int tid = threadIdx.x;
  for (int i = tid; i < 32 * 256; i += 1024) { h_s[i] = 0; c_s[i] = 0.f; }

  const int wave = tid >> 5;
  const int lane = tid & 31;
  const int half = lane >> 4;
  const int nlo  = lane & 15;
  const int jt   = wave & 15;   // column tile within each gate block (H/16 = 16)
  const int mt   = wave >> 4;   // batch tile (0/1)

  const unsigned xgs_base = (unsigned)(size_t)xg_s;
  const int cb = tid >> 5;          // batch row this thread copies
  const int cc = (tid & 31) << 5;   // first of 32 consecutive floats

  // async-copy xg[:, t, :] into LDS buffer `buf` (128B per thread)
  auto issue_xg = [&](int t, int buf) {
    const float* src = xg + ((size_t)cb * T_ + t) * G4H + cc;
    unsigned dst = xgs_base + (unsigned)buf * 131072u +
                   (unsigned)cb * 4096u + (unsigned)(tid & 31) * 128u;
#pragma unroll
    for (int j = 0; j < 8; ++j) {
      asm volatile("global_load_async_to_lds_b128 %0, %1, off"
                   :: "v"(dst + (unsigned)j * 16u),
                      "v"((unsigned long long)(size_t)(src + j * 4))
                   : "memory");
    }
  };

  int lenr[8];
#pragma unroll
  for (int r = 0; r < 8; ++r) lenr[r] = lengths[16 * mt + r + 8 * half];

  issue_xg(0, 0);
  asm volatile("s_wait_asynccnt 0x0" ::: "memory");
  __syncthreads();

  for (int t = 0; t < T_; ++t) {
    if (t + 1 < T_) issue_xg(t + 1, (t + 1) & 1);   // overlaps this whole step

    const float* xgb = xg_s + (size_t)(t & 1) * 32768;
    v8f acc[4];
#pragma unroll
    for (int g = 0; g < 4; ++g) {
      int n0 = (g * 16 + jt) * 16;
#pragma unroll
      for (int r = 0; r < 8; ++r) {
        int b = 16 * mt + r + 8 * half;
        acc[g][r] = xgb[b * G4H + n0 + nlo];        // bias folded in
      }
    }
#pragma unroll
    for (int kk = 0; kk < H_; kk += 32) {
      v16bf a = a_frag(h_s + 16 * mt * 256 + kk, 256);
      int kt = kk >> 5;
#pragma unroll
      for (int g = 0; g < 4; ++g) {
        int nt = g * 16 + jt;
        v16bf b = b_frag_swz(wsz + ((size_t)kt * NT_ + nt) * 512);
        acc[g] = __builtin_amdgcn_wmma_f32_16x16x32_bf16(false, a, false, b,
                                                         (short)0, acc[g], false, false);
      }
    }
    __syncthreads();  // all waves done reading h_s and xg buffer
#pragma unroll
    for (int r = 0; r < 8; ++r) {
      int b  = 16 * mt + r + 8 * half;
      int hc = jt * 16 + nlo;
      float iv = sigm(acc[0][r]);
      float fv = sigm(acc[1][r]);
      float gv = tanhf(acc[2][r]);
      float ov = sigm(acc[3][r]);
      float c_old = c_s[b * 256 + hc];
      float c_new = fv * c_old + iv * gv;
      float h_new = ov * tanhf(c_new);
      float out;
      if (t < lenr[r]) {
        c_s[b * 256 + hc] = c_new;
        h_s[b * 256 + hc] = f2bf(h_new);
        out = h_new;
      } else {
        out = 0.f;
      }
      hout[((size_t)b * T_ + t) * H_ + hc] = out;
    }
    asm volatile("s_wait_asynccnt 0x0" ::: "memory");  // next slab landed
    __syncthreads();  // h updates + xg slab visible before next step
  }
}

// ---- K4: emissions  em = [hf, hb_unrev] @ linW^T + linb  (N=5, VALU) -------

__global__ void __launch_bounds__(256) emission_kernel(
    const float* __restrict__ hf, const float* __restrict__ hb,
    const int* __restrict__ lengths,
    const float* __restrict__ linW, const float* __restrict__ linb,
    float* __restrict__ em) {
  int idx = blockIdx.x * blockDim.x + threadIdx.x;
  if (idx >= BT * K_) return;
  int k  = idx % K_;
  int bt = idx / K_;
  int t  = bt % T_;
  int b  = bt / T_;
  int len = lengths[b];
  int tb = (t < len) ? (len - 1 - t) : t;
  const float* hfp = hf + (size_t)bt * H_;
  const float* hbp = hb + ((size_t)b * T_ + tb) * H_;
  const float* w0  = linW + k * 2 * H_;
  float s = linb[k];
  for (int h = 0; h < H_; ++h) s += hfp[h] * w0[h];
  for (int h = 0; h < H_; ++h) s += hbp[h] * w0[H_ + h];
  em[idx] = s;
}

// ---- K5: CRF llh + Viterbi per batch element -------------------------------

__global__ void crf_kernel(
    const float* __restrict__ em, const int* __restrict__ tags,
    const int* __restrict__ lengths,
    const float* __restrict__ cstart, const float* __restrict__ cend,
    const float* __restrict__ ctrans,
    float* __restrict__ llh, float* __restrict__ preds) {
  __shared__ unsigned char bp[T_][K_];
  int b = blockIdx.x;
  if (threadIdx.x != 0) return;

  const float* emb = em + (size_t)b * T_ * K_;
  const int*   tg  = tags + b * T_;
  int len = lengths[b];

  float trans[K_][K_];
  for (int i = 0; i < K_; ++i)
    for (int j = 0; j < K_; ++j) trans[i][j] = ctrans[i * K_ + j];

  // gold path score
  float score = cstart[tg[0]] + emb[tg[0]];
  for (int t = 1; t < len; ++t)
    score += trans[tg[t - 1]][tg[t]] + emb[t * K_ + tg[t]];
  score += cend[tg[len - 1]];

  // forward algorithm (masked steps leave alpha unchanged -> stop at len)
  float alpha[K_];
  for (int j = 0; j < K_; ++j) alpha[j] = cstart[j] + emb[j];
  for (int t = 1; t < len; ++t) {
    float na[K_];
    for (int j = 0; j < K_; ++j) {
      float mx = -1e30f;
      for (int i = 0; i < K_; ++i) { float v = alpha[i] + trans[i][j]; if (v > mx) mx = v; }
      float s = 0.f;
      for (int i = 0; i < K_; ++i) s += expf(alpha[i] + trans[i][j] - mx);
      na[j] = mx + logf(s) + emb[t * K_ + j];
    }
    for (int j = 0; j < K_; ++j) alpha[j] = na[j];
  }
  float mx = -1e30f;
  for (int j = 0; j < K_; ++j) { float v = alpha[j] + cend[j]; if (v > mx) mx = v; }
  float se = 0.f;
  for (int j = 0; j < K_; ++j) se += expf(alpha[j] + cend[j] - mx);
  llh[b] = score - (mx + logf(se));

  // Viterbi
  float vs[K_];
  for (int j = 0; j < K_; ++j) vs[j] = cstart[j] + emb[j];
  for (int t = 1; t < T_; ++t) {
    if (t < len) {
      float nv[K_];
      for (int j = 0; j < K_; ++j) {
        float best = -1e30f; int arg = 0;
        for (int i = 0; i < K_; ++i) {
          float v = vs[i] + trans[i][j];
          if (v > best) { best = v; arg = i; }
        }
        nv[j] = best + emb[t * K_ + j];
        bp[t][j] = (unsigned char)arg;
      }
      for (int j = 0; j < K_; ++j) vs[j] = nv[j];
    } else {
      for (int j = 0; j < K_; ++j) bp[t][j] = (unsigned char)j;  // identity
    }
  }
  int cur = 0;
  { float best = -1e30f;
    for (int j = 0; j < K_; ++j) { float v = vs[j] + cend[j]; if (v > best) { best = v; cur = j; } } }
  for (int t = T_ - 1; t >= 1; --t) {
    preds[(size_t)b * T_ + t] = (t < len) ? (float)cur : 0.f;
    cur = bp[t][cur];
  }
  preds[(size_t)b * T_ + 0] = (float)cur;  // len >= 1 always
}

// ---- K6: deterministic loss reduction --------------------------------------

__global__ void finalize_kernel(const float* __restrict__ llh, float* __restrict__ out0) {
  if (threadIdx.x == 0 && blockIdx.x == 0) {
    float s = 0.f;
    for (int b = 0; b < B_; ++b) s += llh[b];
    out0[0] = -s / (float)B_;
  }
}

// ---- host ------------------------------------------------------------------

extern "C" void kernel_launch(void* const* d_in, const int* in_sizes, int n_in,
                              void* d_out, int out_size, void* d_ws, size_t ws_size,
                              hipStream_t stream) {
  const float* we      = (const float*)d_in[0];
  const int*   lengths = (const int*)d_in[1];
  // d_in[2] token_mask: derived from lengths instead (dtype-safe)
  const int*   tags    = (const int*)d_in[3];
  const float* wihf    = (const float*)d_in[4];
  const float* whhf    = (const float*)d_in[5];
  const float* bf      = (const float*)d_in[6];
  const float* wihb    = (const float*)d_in[7];
  const float* whhb    = (const float*)d_in[8];
  const float* bb      = (const float*)d_in[9];
  const float* linW    = (const float*)d_in[10];
  const float* linb    = (const float*)d_in[11];
  const float* cstart  = (const float*)d_in[12];
  const float* cend    = (const float*)d_in[13];
  const float* ctrans  = (const float*)d_in[14];

  char* ws = (char*)d_ws;
  size_t off = 0;
  auto alloc = [&](size_t bytes) -> void* {
    void* p = ws + off;
    off += (bytes + 255) & ~(size_t)255;
    return p;
  };
  unsigned short* embs_f  = (unsigned short*)alloc((size_t)BT * E_ * 2);
  unsigned short* embs_r  = (unsigned short*)alloc((size_t)BT * E_ * 2);
  unsigned short* wihf_s  = (unsigned short*)alloc((size_t)E_ * G4H * 2);
  unsigned short* wihb_s  = (unsigned short*)alloc((size_t)E_ * G4H * 2);
  unsigned short* whhf_s  = (unsigned short*)alloc((size_t)H_ * G4H * 2);
  unsigned short* whhb_s  = (unsigned short*)alloc((size_t)H_ * G4H * 2);
  float*          xg_f    = (float*)alloc((size_t)BT * G4H * 4);
  float*          xg_b    = (float*)alloc((size_t)BT * G4H * 4);
  float*          hf      = (float*)alloc((size_t)BT * H_ * 4);
  float*          hb      = (float*)alloc((size_t)BT * H_ * 4);
  float*          em      = (float*)alloc((size_t)BT * K_ * 4);
  float*          llh     = (float*)alloc(B_ * 4);

  float* loss_out  = (float*)d_out;
  float* preds_out = (float*)d_out + 1;

  pack_weights_kernel<<<dim3((G4H * E_ + 255) / 256), 256, 0, stream>>>(
      wihf, wihb, whhf, whhb, wihf_s, wihb_s, whhf_s, whhb_s);

  pack_embs_kernel<<<dim3((unsigned)(((size_t)BT * E_ + 255) / 256)), 256, 0, stream>>>(
      we, lengths, embs_f, embs_r);

  xg_gemm_kernel<<<dim3(BT / 16, 2, 2), 256, 0, stream>>>(
      embs_f, embs_r, wihf_s, wihb_s, bf, bb, xg_f, xg_b);

  // dynamic LDS: 2 x 32x1024 f32 = 256KB (+48KB static = 304KB of 320KB WGP)
  lstm_kernel<<<dim3(2), 1024, 262144, stream>>>(
      xg_f, xg_b, whhf_s, whhb_s, lengths, hf, hb);

  emission_kernel<<<dim3((BT * K_ + 255) / 256), 256, 0, stream>>>(
      hf, hb, lengths, linW, linb, em);

  crf_kernel<<<dim3(B_), 32, 0, stream>>>(
      em, tags, lengths, cstart, cend, ctrans, llh, preds_out);

  finalize_kernel<<<1, 32, 0, stream>>>(llh, loss_out);
}
